// Decoder_11587821764937
// MI455X (gfx1250) — matrix operand order repaired and checked by reading, not compiled, not run on hardware
//
#include <hip/hip_runtime.h>
#include <hip/hip_bf16.h>
#include <stdint.h>

// ---------------- problem constants ----------------
#define BB      32
#define T_IN    400
#define T_OUTX  500
#define N_MEL   80
#define N_MELP  96          // padded to multiple of 32 for WMMA K
#define ENC     768
#define ARNN    1024
#define DRNN    1024
#define PREN    256
#define ATTD    128
#define NFILT   32
#define KSZ     31
#define TOKD    128
#define GO_OFF  (-13.815f)

typedef __attribute__((ext_vector_type(16))) __bf16 v16bf;
typedef __attribute__((ext_vector_type(8)))  __bf16 v8bf;
typedef __attribute__((ext_vector_type(8)))  float  v8f;

// ---------------- small helpers ----------------
__device__ __forceinline__ unsigned hash32(unsigned x, unsigned seed) {
    x ^= seed * 0x9E3779B9u;
    x *= 0x85EBCA6Bu; x ^= x >> 13;
    x *= 0xC2B2AE35u; x ^= x >> 16;
    return x;
}
__device__ __forceinline__ float apply_dropout(float v, float keep, unsigned seed, unsigned idx) {
    unsigned h = hash32(idx, seed);
    float r = (float)(h >> 8) * (1.0f / 16777216.0f);
    return (r < keep) ? (v / keep) : 0.0f;
}
__device__ __forceinline__ float sigm(float x) { return 1.0f / (1.0f + __expf(-x)); }

// ---------------- utility kernels ----------------
__global__ void k_fill(float* p, float v, int n) {
    int i = blockIdx.x * blockDim.x + threadIdx.x;
    if (i < n) p[i] = v;
}
__global__ void k_fill_bf(__bf16* p, float v, int n) {
    int i = blockIdx.x * blockDim.x + threadIdx.x;
    if (i < n) p[i] = (__bf16)v;
}
__global__ void k_cvt_bf16(const float* __restrict__ src, __bf16* __restrict__ dst, int n) {
    int i = blockIdx.x * blockDim.x + threadIdx.x;
    if (i < n) dst[i] = (__bf16)src[i];
}
// convert (N x Ks) fp32 -> (N x Kd) bf16, zero-padded columns Ks..Kd-1
__global__ void k_cvt_pad(const float* __restrict__ src, __bf16* __restrict__ dst,
                          int N, int Ks, int Kd) {
    int i = blockIdx.x * blockDim.x + threadIdx.x;
    if (i >= N * Kd) return;
    int n = i / Kd, k = i % Kd;
    dst[i] = (k < Ks) ? (__bf16)src[n * Ks + k] : (__bf16)0.0f;
}
// gather x_t (B x N_MELP) as bf16: teacher-forced frame t-1 (or GO), zero-padded
__global__ void k_gather_x(const float* __restrict__ dec_inputs, __bf16* __restrict__ x, int t) {
    int i = blockIdx.x * blockDim.x + threadIdx.x;
    if (i >= BB * N_MELP) return;
    int b = i / N_MELP, k = i % N_MELP;
    float v = 0.0f;
    if (k < N_MEL) v = (t == 0) ? GO_OFF
                                : dec_inputs[(size_t)b * N_MEL * T_OUTX + (size_t)k * T_OUTX + (t - 1)];
    x[i] = (__bf16)v;
}

// ---------------- generic 3-input WMMA GEMM (bf16 x bf16 -> fp32) -----------
// out[M,N] = sum_i A_i[M,K_i] * W_i[N,K_i]^T (+bias0[n] +bias1[n])
// Requirements: K % 32 == 0; A,W bf16 with rows contiguous (ld mult of 8
// elements); M,N multiples of 16; base pointers 16B aligned.
struct GemmIn {
    const __bf16* A;
    int           ldA;
    const __bf16* W;
    int           ldW;
    int           K;
};

__device__ __forceinline__ void gemm_accum(v8f& acc, const GemmIn g,
                                           int row0, int col0, int gl, int il) {
    const __bf16* Arow = g.A + (size_t)(row0 + il) * g.ldA;
    const __bf16* Wrow = g.W + (size_t)(col0 + il) * g.ldW;
    for (int k0 = 0; k0 < g.K; k0 += 32) {
        // A 16x32 bf16 fragment: per lane 16 contiguous bf16 at k0+16*gl
        // (same k order as B).
        v16bf a = *(const v16bf*)(Arow + k0 + 16 * gl);
        // B 32x16 fragment: per lane 16 contiguous bf16 at k0+16*gl.
        v16bf b = *(const v16bf*)(Wrow + k0 + 16 * gl);
        acc = __builtin_amdgcn_wmma_f32_16x16x32_bf16(
            false, a, false, b, (short)0, acc, false, false);
    }
}

__global__ void k_wmma_gemm3(GemmIn g0, GemmIn g1, GemmIn g2,
                             const float* bias0, const float* bias1,
                             float* out, __bf16* outBf, int M, int N,
                             int ldOut, int epi, float keep, unsigned seed) {
    int wave = threadIdx.x >> 5;
    int lane = threadIdx.x & 31;
    int ntN  = N >> 4;
    int tile = blockIdx.x * (blockDim.x >> 5) + wave;
    int ntot = (M >> 4) * ntN;
    if (tile >= ntot) return;          // wave-uniform
    int row0 = (tile / ntN) << 4;
    int col0 = (tile % ntN) << 4;
    int gl = lane >> 4;
    int il = lane & 15;

    v8f acc = {};
    gemm_accum(acc, g0, row0, col0, gl, il);
    gemm_accum(acc, g1, row0, col0, gl, il);
    gemm_accum(acc, g2, row0, col0, gl, il);

#pragma unroll
    for (int r = 0; r < 8; ++r) {
        int m = row0 + r + 8 * gl;     // C/D layout: VGPR r -> M=r (+8 for hi lanes)
        int n = col0 + il;
        float v = acc[r];
        if (bias0) v += bias0[n];
        if (bias1) v += bias1[n];
        if (epi == 1) {                // relu + dropout
            v = fmaxf(v, 0.0f);
            v = apply_dropout(v, keep, seed, (unsigned)(m * N + n));
        }
        out[(size_t)m * ldOut + n] = v;
        if (outBf) outBf[(size_t)m * ldOut + n] = (__bf16)v;
    }
}

// ---------------- LSTM cell (elementwise) ----------------
__global__ void k_lstm_cell(const float* __restrict__ gates, float* h, __bf16* h_bf,
                            float* c, int nh, float keep, unsigned seed) {
    int idx = blockIdx.x * blockDim.x + threadIdx.x;
    if (idx >= BB * nh) return;
    int b = idx / nh, n = idx % nh;
    const float* g = gates + (size_t)b * 4 * nh;
    float ig = g[n], fg = g[n + nh], gg = g[n + 2 * nh], og = g[n + 3 * nh];
    float c2 = sigm(fg) * c[idx] + sigm(ig) * tanhf(gg);
    float hv = sigm(og) * tanhf(c2);
    hv = apply_dropout(hv, keep, seed, (unsigned)idx);
    c[idx] = c2;
    h[idx] = hv;
    h_bf[idx] = (__bf16)hv;
}

// ---------------- location-sensitive attention (1 block per batch row) ------
__global__ void k_attention(const float* __restrict__ memory,
                            const float* __restrict__ pm,     // (B*T_IN, ATTD)
                            const float* __restrict__ pq,     // (B, ATTD)
                            const unsigned char* __restrict__ mask,
                            const float* __restrict__ lc,     // (NF,2,KS)
                            const float* __restrict__ ldw,    // (ATTD,NF)
                            const float* __restrict__ vw,     // (ATTD)
                            float* att_w, float* att_w_cum,
                            float* ctx, __bf16* ctx_bf,
                            float* align_out, int t) {
    __shared__ float s_w[T_IN], s_cum[T_IN], s_e[T_IN], s_red[256];
    __shared__ float s_lc[NFILT * 2 * KSZ], s_ld[ATTD * NFILT], s_v[ATTD], s_pq[ATTD];
    int b = blockIdx.x, tid = threadIdx.x, nt = blockDim.x;

    for (int i = tid; i < NFILT * 2 * KSZ; i += nt) s_lc[i] = lc[i];
    for (int i = tid; i < ATTD * NFILT;   i += nt) s_ld[i] = ldw[i];
    for (int i = tid; i < ATTD; i += nt) { s_v[i] = vw[i]; s_pq[i] = pq[b * ATTD + i]; }
    for (int i = tid; i < T_IN; i += nt) {
        s_w[i]   = att_w[b * T_IN + i];
        s_cum[i] = att_w_cum[b * T_IN + i];
    }
    __syncthreads();

    float lmax = -3.4e38f;
    for (int ti = tid; ti < T_IN; ti += nt) {
        float conv[NFILT];
#pragma unroll
        for (int f = 0; f < NFILT; ++f) conv[f] = 0.0f;
        for (int k = 0; k < KSZ; ++k) {
            int tt = ti + k - KSZ / 2;
            float c0 = 0.0f, c1 = 0.0f;
            if (tt >= 0 && tt < T_IN) { c0 = s_w[tt]; c1 = s_cum[tt]; }
#pragma unroll
            for (int f = 0; f < NFILT; ++f)
                conv[f] += c0 * s_lc[f * 62 + k] + c1 * s_lc[f * 62 + 31 + k];
        }
        float e = 0.0f;
        const float* pmr = pm + ((size_t)b * T_IN + ti) * ATTD;
        for (int a = 0; a < ATTD; ++a) {
            float s = s_pq[a] + pmr[a];
#pragma unroll
            for (int f = 0; f < NFILT; ++f) s += conv[f] * s_ld[a * NFILT + f];
            e += tanhf(s) * s_v[a];
        }
        if (mask[b * T_IN + ti]) e = -1e9f;
        s_e[ti] = e;
        lmax = fmaxf(lmax, e);
    }
    s_red[tid] = lmax; __syncthreads();
    for (int s = nt >> 1; s > 0; s >>= 1) {
        if (tid < s) s_red[tid] = fmaxf(s_red[tid], s_red[tid + s]);
        __syncthreads();
    }
    float gmax = s_red[0]; __syncthreads();

    float lsum = 0.0f;
    for (int ti = tid; ti < T_IN; ti += nt) {
        float w = __expf(s_e[ti] - gmax);
        s_e[ti] = w;
        lsum += w;
    }
    s_red[tid] = lsum; __syncthreads();
    for (int s = nt >> 1; s > 0; s >>= 1) {
        if (tid < s) s_red[tid] += s_red[tid + s];
        __syncthreads();
    }
    float inv = 1.0f / s_red[0]; __syncthreads();

    for (int ti = tid; ti < T_IN; ti += nt) {
        float w = s_e[ti] * inv;
        s_e[ti] = w;
        att_w[b * T_IN + ti]     = w;
        att_w_cum[b * T_IN + ti] = s_cum[ti] + w;
        align_out[((size_t)b * T_OUTX + t) * T_IN + ti] = w;
    }
    __syncthreads();

    for (int e0 = tid; e0 < ENC; e0 += nt) {
        float s = 0.0f;
        const float* mrow = memory + (size_t)b * T_IN * ENC + e0;
        for (int tt = 0; tt < T_IN; ++tt) s += s_e[tt] * mrow[(size_t)tt * ENC];
        ctx[b * ENC + e0] = s;
        ctx_bf[b * ENC + e0] = (__bf16)s;
    }
}

// ---------------- gate + mel scatter ----------------
__global__ void k_finalize(const float* __restrict__ dec_h, const float* __restrict__ ctx,
                           const float* __restrict__ gw, const float* __restrict__ gb,
                           const float* __restrict__ mel_ws,
                           float* out_mel, float* out_gate, int t) {
    __shared__ float red[256];
    int b = blockIdx.x, tid = threadIdx.x;
    float p = 0.0f;
    for (int k = tid; k < DRNN + ENC; k += blockDim.x) {
        float x = (k < DRNN) ? dec_h[b * DRNN + k] : ctx[b * ENC + (k - DRNN)];
        p += x * gw[k];
    }
    red[tid] = p; __syncthreads();
    for (int s = blockDim.x >> 1; s > 0; s >>= 1) {
        if (tid < s) red[tid] += red[tid + s];
        __syncthreads();
    }
    if (tid == 0) out_gate[b * T_OUTX + t] = red[0] + gb[0];
    for (int n = tid; n < N_MEL; n += blockDim.x)
        out_mel[(size_t)b * N_MEL * T_OUTX + (size_t)n * T_OUTX + t] = mel_ws[b * N_MEL + n];
}

// ---------------- initial states from GST ----------------
__global__ void k_init_states(const float* __restrict__ gst,
                              const float* __restrict__ W1,   // (4,8,TOK)
                              const float* __restrict__ W2,   // (4,ARNN,8)
                              float* att_h, float* att_c, float* dec_h, float* dec_c,
                              __bf16* att_h_bf, __bf16* dec_h_bf) {
    __shared__ float h8[8];
    int i = blockIdx.x / BB, b = blockIdx.x % BB, tid = threadIdx.x;
    if (tid < 8) {
        float s = 0.0f;
        for (int k = 0; k < TOKD; ++k) s += gst[b * TOKD + k] * W1[(i * 8 + tid) * TOKD + k];
        s = fmaxf(s, 0.0f);
        s = apply_dropout(s, 0.5f, 0x51000u + 2 * i, (unsigned)(b * 8 + tid));
        h8[tid] = s;
    }
    __syncthreads();
    float*  dst   = (i == 0) ? att_h : (i == 1) ? att_c : (i == 2) ? dec_h : dec_c;
    __bf16* dstbf = (i == 0) ? att_h_bf : (i == 2) ? dec_h_bf : nullptr;
    for (int n = tid; n < ARNN; n += blockDim.x) {
        float s = 0.0f;
#pragma unroll
        for (int j = 0; j < 8; ++j) s += h8[j] * W2[((size_t)i * ARNN + n) * 8 + j];
        s = fmaxf(s, 0.0f);
        s = apply_dropout(s, 0.5f, 0x51001u + 2 * i, (unsigned)(b * ARNN + n));
        dst[b * ARNN + n] = s;
        if (dstbf) dstbf[b * ARNN + n] = (__bf16)s;
    }
}

// ---------------- host side ----------------
static inline int gemm_blocks(int M, int N) {
    int tiles = (M / 16) * (N / 16);
    return (tiles + 7) / 8;
}

extern "C" void kernel_launch(void* const* d_in, const int* in_sizes, int n_in,
                              void* d_out, int out_size, void* d_ws, size_t ws_size,
                              hipStream_t stream) {
    (void)in_sizes; (void)n_in; (void)out_size; (void)ws_size;
    const float* memory     = (const float*)d_in[0];
    const float* dec_inputs = (const float*)d_in[1];
    const float* gst        = (const float*)d_in[2];
    const unsigned char* mask = (const unsigned char*)d_in[3];
    const float* prenet_W1  = (const float*)d_in[4];
    const float* prenet_W2  = (const float*)d_in[5];
    const float* att_Wih    = (const float*)d_in[6];
    const float* att_Whh    = (const float*)d_in[7];
    const float* att_bih    = (const float*)d_in[8];
    const float* att_bhh    = (const float*)d_in[9];
    const float* q_W        = (const float*)d_in[10];
    const float* mem_W      = (const float*)d_in[11];
    const float* v_W        = (const float*)d_in[12];
    const float* loc_conv_W = (const float*)d_in[13];
    const float* loc_dense_W= (const float*)d_in[14];
    const float* dec_Wih    = (const float*)d_in[15];
    const float* dec_Whh    = (const float*)d_in[16];
    const float* dec_bih    = (const float*)d_in[17];
    const float* dec_bhh    = (const float*)d_in[18];
    const float* proj_W     = (const float*)d_in[19];
    const float* proj_b     = (const float*)d_in[20];
    const float* gate_W     = (const float*)d_in[21];
    const float* gate_b     = (const float*)d_in[22];
    const float* init_W1    = (const float*)d_in[23];
    const float* init_W2    = (const float*)d_in[24];

    float* out_mel   = (float*)d_out;
    float* out_gate  = out_mel + (size_t)BB * N_MEL * T_OUTX;
    float* out_align = out_gate + (size_t)BB * T_OUTX;

    // ---- carve workspace (256B aligned blocks) ----
    char* base = (char*)d_ws;
    size_t off = 0;
    auto alloc = [&](size_t bytes) -> void* {
        void* p = base + off;
        off += (bytes + 255) & ~(size_t)255;
        return p;
    };
    __bf16* bf_pw1  = (__bf16*)alloc((size_t)PREN * N_MELP * 2);   // padded K
    __bf16* bf_pw2  = (__bf16*)alloc((size_t)PREN * PREN * 2);
    __bf16* bf_aWih = (__bf16*)alloc((size_t)4 * ARNN * (PREN + ENC) * 2);
    __bf16* bf_aWhh = (__bf16*)alloc((size_t)4 * ARNN * ARNN * 2);
    __bf16* bf_qW   = (__bf16*)alloc((size_t)ATTD * ARNN * 2);
    __bf16* bf_mW   = (__bf16*)alloc((size_t)ATTD * ENC * 2);
    __bf16* bf_dWih = (__bf16*)alloc((size_t)4 * DRNN * (ARNN + ENC) * 2);
    __bf16* bf_dWhh = (__bf16*)alloc((size_t)4 * DRNN * DRNN * 2);
    __bf16* bf_pjW  = (__bf16*)alloc((size_t)N_MEL * (DRNN + ENC) * 2);
    __bf16* bf_mem  = (__bf16*)alloc((size_t)BB * T_IN * ENC * 2);
    __bf16* bf_ah   = (__bf16*)alloc((size_t)BB * ARNN * 2);
    __bf16* bf_dh   = (__bf16*)alloc((size_t)BB * DRNN * 2);
    __bf16* bf_ctx  = (__bf16*)alloc((size_t)BB * ENC * 2);
    __bf16* bf_x    = (__bf16*)alloc((size_t)BB * N_MELP * 2);
    __bf16* bf_p1   = (__bf16*)alloc((size_t)BB * PREN * 2);
    __bf16* bf_p2   = (__bf16*)alloc((size_t)BB * PREN * 2);
    float* f_pm    = (float*)alloc((size_t)BB * T_IN * ATTD * 4);
    float* f_ah    = (float*)alloc((size_t)BB * ARNN * 4);
    float* f_ac    = (float*)alloc((size_t)BB * ARNN * 4);
    float* f_dh    = (float*)alloc((size_t)BB * DRNN * 4);
    float* f_dc    = (float*)alloc((size_t)BB * DRNN * 4);
    float* f_aw    = (float*)alloc((size_t)BB * T_IN * 4);
    float* f_awc   = (float*)alloc((size_t)BB * T_IN * 4);
    float* f_ctx   = (float*)alloc((size_t)BB * ENC * 4);
    float* f_p1    = (float*)alloc((size_t)BB * PREN * 4);
    float* f_p2    = (float*)alloc((size_t)BB * PREN * 4);
    float* f_gates = (float*)alloc((size_t)BB * 4 * DRNN * 4);
    float* f_pq    = (float*)alloc((size_t)BB * ATTD * 4);
    float* f_mel   = (float*)alloc((size_t)BB * N_MEL * 4);

    auto cvt = [&](const float* s, __bf16* d, int n) {
        k_cvt_bf16<<<(n + 255) / 256, 256, 0, stream>>>(s, d, n);
    };
    k_cvt_pad<<<(PREN * N_MELP + 255) / 256, 256, 0, stream>>>(
        prenet_W1, bf_pw1, PREN, N_MEL, N_MELP);
    cvt(prenet_W2, bf_pw2, PREN * PREN);
    cvt(att_Wih, bf_aWih, 4 * ARNN * (PREN + ENC));
    cvt(att_Whh, bf_aWhh, 4 * ARNN * ARNN);
    cvt(q_W, bf_qW, ATTD * ARNN);
    cvt(mem_W, bf_mW, ATTD * ENC);
    cvt(dec_Wih, bf_dWih, 4 * DRNN * (ARNN + ENC));
    cvt(dec_Whh, bf_dWhh, 4 * DRNN * DRNN);
    cvt(proj_W, bf_pjW, N_MEL * (DRNN + ENC));
    cvt(memory, bf_mem, BB * T_IN * ENC);

    k_fill<<<(BB * T_IN + 255) / 256, 256, 0, stream>>>(f_aw, 0.0f, BB * T_IN);
    k_fill<<<(BB * T_IN + 255) / 256, 256, 0, stream>>>(f_awc, 0.0f, BB * T_IN);
    k_fill<<<(BB * ENC + 255) / 256, 256, 0, stream>>>(f_ctx, 0.0f, BB * ENC);
    k_fill_bf<<<(BB * ENC + 255) / 256, 256, 0, stream>>>(bf_ctx, 0.0f, BB * ENC);

    k_init_states<<<4 * BB, 256, 0, stream>>>(gst, init_W1, init_W2,
                                              f_ah, f_ac, f_dh, f_dc, bf_ah, bf_dh);

    GemmIn Z = {nullptr, 0, nullptr, 0, 0};

    // processed_memory = memory @ mem_W^T : (B*T_IN, ENC) x (ATTD, ENC)^T
    {
        GemmIn g0 = {bf_mem, ENC, bf_mW, ENC, ENC};
        k_wmma_gemm3<<<gemm_blocks(BB * T_IN, ATTD), 256, 0, stream>>>(
            g0, Z, Z, nullptr, nullptr, f_pm, nullptr, BB * T_IN, ATTD, ATTD, 0, 1.0f, 0);
    }

    for (int t = 0; t < T_OUTX; ++t) {
        unsigned seed = 0xC0FFEEu + (unsigned)t * 16u;

        // gather teacher-forced frame (bf16, zero-padded to K=96)
        k_gather_x<<<(BB * N_MELP + 255) / 256, 256, 0, stream>>>(dec_inputs, bf_x, t);

        // prenet layer 1: (32x96)x(256x96)^T, relu+dropout(0.5)
        {
            GemmIn g0 = {bf_x, N_MELP, bf_pw1, N_MELP, N_MELP};
            k_wmma_gemm3<<<gemm_blocks(BB, PREN), 256, 0, stream>>>(
                g0, Z, Z, nullptr, nullptr, f_p1, bf_p1, BB, PREN, PREN, 1, 0.5f, seed + 1);
        }
        // prenet layer 2
        {
            GemmIn g0 = {bf_p1, PREN, bf_pw2, PREN, PREN};
            k_wmma_gemm3<<<gemm_blocks(BB, PREN), 256, 0, stream>>>(
                g0, Z, Z, nullptr, nullptr, f_p2, bf_p2, BB, PREN, PREN, 1, 0.5f, seed + 2);
        }
        // attention LSTM gates: [p2 | ctx] @ Wih^T + att_h @ Whh^T + biases
        {
            GemmIn g0 = {bf_p2, PREN, bf_aWih, PREN + ENC, PREN};
            GemmIn g1 = {bf_ctx, ENC, bf_aWih + PREN, PREN + ENC, ENC};
            GemmIn g2 = {bf_ah, ARNN, bf_aWhh, ARNN, ARNN};
            k_wmma_gemm3<<<gemm_blocks(BB, 4 * ARNN), 256, 0, stream>>>(
                g0, g1, g2, att_bih, att_bhh, f_gates, nullptr, BB, 4 * ARNN, 4 * ARNN,
                0, 1.0f, 0);
        }
        k_lstm_cell<<<(BB * ARNN + 255) / 256, 256, 0, stream>>>(
            f_gates, f_ah, bf_ah, f_ac, ARNN, 0.9f, seed + 3);

        // query projection pq = att_h @ q_W^T
        {
            GemmIn g0 = {bf_ah, ARNN, bf_qW, ARNN, ARNN};
            k_wmma_gemm3<<<gemm_blocks(BB, ATTD), 256, 0, stream>>>(
                g0, Z, Z, nullptr, nullptr, f_pq, nullptr, BB, ATTD, ATTD, 0, 1.0f, 0);
        }
        // location-sensitive attention + context
        k_attention<<<BB, 256, 0, stream>>>(memory, f_pm, f_pq, mask,
                                            loc_conv_W, loc_dense_W, v_W,
                                            f_aw, f_awc, f_ctx, bf_ctx, out_align, t);
        // decoder LSTM gates: [att_h | ctx] @ dec_Wih^T + dec_h @ dec_Whh^T + biases
        {
            GemmIn g0 = {bf_ah, ARNN, bf_dWih, ARNN + ENC, ARNN};
            GemmIn g1 = {bf_ctx, ENC, bf_dWih + ARNN, ARNN + ENC, ENC};
            GemmIn g2 = {bf_dh, DRNN, bf_dWhh, DRNN, DRNN};
            k_wmma_gemm3<<<gemm_blocks(BB, 4 * DRNN), 256, 0, stream>>>(
                g0, g1, g2, dec_bih, dec_bhh, f_gates, nullptr, BB, 4 * DRNN, 4 * DRNN,
                0, 1.0f, 0);
        }
        k_lstm_cell<<<(BB * DRNN + 255) / 256, 256, 0, stream>>>(
            f_gates, f_dh, bf_dh, f_dc, DRNN, 0.9f, seed + 4);

        // mel projection: [dec_h | ctx] @ proj_W^T + proj_b  (N = 80 = 5 tiles)
        {
            GemmIn g0 = {bf_dh, DRNN, bf_pjW, DRNN + ENC, DRNN};
            GemmIn g1 = {bf_ctx, ENC, bf_pjW + DRNN, DRNN + ENC, ENC};
            k_wmma_gemm3<<<gemm_blocks(BB, N_MEL), 256, 0, stream>>>(
                g0, g1, Z, proj_b, nullptr, f_mel, nullptr, BB, N_MEL, N_MEL, 0, 1.0f, 0);
        }
        // gate scalar + mel scatter to output layout
        k_finalize<<<BB, 256, 0, stream>>>(f_dh, f_ctx, gate_W, gate_b,
                                           f_mel, out_mel, out_gate, t);
    }
}
